// ResNetNLM_29772713296446
// MI455X (gfx1250) — matrix operand order repaired and checked by reading, not compiled, not run on hardware
//
#include <hip/hip_runtime.h>

typedef __attribute__((ext_vector_type(16))) _Float16 v16h;
typedef __attribute__((ext_vector_type(8)))  float    v8f;

#define Bsz     8
#define Nn      256
#define CIMG    2048
#define Hdim    128
#define NODE_C  14
#define EDGE_C  10
#define PAIRS_PER_B (Nn * (Nn - 1))          // 65280
#define NPAIR       (Bsz * PAIRS_PER_B)      // 522240
#define NODE_OUT    (Bsz * Nn * NODE_C)      // 28672
#define EDGE_ATTR_OUT (NPAIR * (EDGE_C + 1)) // 5744640
#define DENSE_OUT   (Bsz * Nn * Nn * (EDGE_C + 1)) // 5767168
#define XCHUNK  512                           // K-chunk for node layer 1

// --- WMMA f16 fragment maps (CDNA5 ISA 7.12.2, wave32) ---
// A 16x32 f16: lane holds row m=lane&15; element e -> K:
__device__ __forceinline__ int ak_idx(int lane, int e) {
    return e + ((e >= 8) ? 8 : 0) + ((lane >= 16) ? 8 : 0);
}
// B 32x16 f16: lane holds col n=lane&15; element e -> K:
__device__ __forceinline__ int bk_idx(int lane, int e) {
    return e + ((e >= 8) ? 8 : 0) + ((lane >= 16) ? 8 : 0);
}
// Inverse of A map: producer holding (row mr, col nc) of an activation tile
// finds the fragment slot (for row 0); row mr adds mr*16.
__device__ __forceinline__ int frag_slot0(int nc) {
    int kt = nc >> 5, kin = nc & 31;
    int half, e;
    if (kin < 8)       { half = 0; e = kin;      }
    else if (kin < 16) { half = 1; e = kin - 8;  }
    else if (kin < 24) { half = 0; e = kin - 8;  }
    else               { half = 1; e = kin - 16; }
    return ((kt * 32 + half * 16) << 4) + e;
}
__device__ __forceinline__ float sigmoidf_(float x) {
    return 1.0f / (1.0f + __expf(-x));
}
__device__ __forceinline__ v8f wmma16(v16h a, v16h b, v8f c) {
    return __builtin_amdgcn_wmma_f32_16x16x32_f16(false, a, false, b,
                                                  (short)0, c, false, false);
}

__global__ __launch_bounds__(256) void zero_f4_kernel(float4* __restrict__ p, int n4) {
    int i = blockIdx.x * blockDim.x + threadIdx.x;
    if (i < n4) p[i] = make_float4(0.f, 0.f, 0.f, 0.f);
}

// Pack W[Ksrc x Nsrc] (f32, row-major K x N) into f16 B-fragments:
// dst[((kt*ntiles + nt)*32 + lane)*16 + e] = W[kt*32 + bk(lane,e)][nt*16 + lane%16]
// Zero-pads K beyond Ksrc and N beyond Nsrc.
__global__ __launch_bounds__(256) void pack_w_kernel(
    const float* __restrict__ W, int Ksrc, int Nsrc, int ktiles, int ntiles,
    _Float16* __restrict__ dst)
{
    int idx = blockIdx.x * 256 + threadIdx.x;
    int total = ktiles * ntiles * 512;
    if (idx >= total) return;
    int e    = idx & 15;
    int lane = (idx >> 4) & 31;
    int frag = idx >> 9;
    int nt = frag % ntiles, kt = frag / ntiles;
    int k = kt * 32 + bk_idx(lane, e);
    int n = nt * 16 + (lane & 15);
    float v = (k < Ksrc && n < Nsrc) ? W[k * Nsrc + n] : 0.f;
    dst[idx] = (_Float16)v;
}

// ---------------- Node MLP: 2048 -> relu128 -> relu128 -> sigmoid14 ----------------
__global__ __launch_bounds__(256) void node_mlp_kernel(
    const float* __restrict__ roi,
    const _Float16* __restrict__ W1p, const float* __restrict__ b1,
    const _Float16* __restrict__ W2p, const float* __restrict__ b2,
    const _Float16* __restrict__ W3p, const float* __restrict__ b3,
    float* __restrict__ out_node)
{
    __shared__ _Float16 lds_xA[16 * XCHUNK];  // 16 K-tiles of A fragments (16KB)
    __shared__ _Float16 lds_h1A[16 * Hdim];   // 4 K-tiles, fragment layout (4KB)
    __shared__ _Float16 lds_h2A[16 * Hdim];

    const int row0 = blockIdx.x * 16;
    const int tid  = threadIdx.x;
    const int lane = tid & 31;
    const int wave = tid >> 5;                // 0..7 -> N tile
    const int nc   = wave * 16 + (lane & 15);
    const int sb   = frag_slot0(nc);          // producer store base
    const int radd = (lane >= 16) ? 8 : 0;

    // ---- layer 1: K=2048 streamed in fragment layout ----
    v8f acc = {};
    for (int c0 = 0; c0 < CIMG; c0 += XCHUNK) {
        for (int s = tid; s < 16 * XCHUNK; s += 256) {
            int e = s & 15, l = (s >> 4) & 31, kt = s >> 9;
            int k = c0 + kt * 32 + ak_idx(l, e);
            lds_xA[s] = (_Float16)roi[(row0 + (l & 15)) * CIMG + k];
        }
        __syncthreads();
        const _Float16* wbase = &W1p[(((c0 >> 5) * 8 + wave) * 32 + lane) << 4];
        for (int kt = 0; kt < XCHUNK / 32; ++kt) {
            v16h a  = *(const v16h*)&lds_xA[(kt * 32 + lane) << 4];
            v16h bm = *(const v16h*)&wbase[kt * (8 * 512)];
            acc = wmma16(a, bm, acc);
        }
        __syncthreads();
    }
    {
        float bias = b1[nc];
        #pragma unroll
        for (int r = 0; r < 8; ++r) {
            float v = acc[r] + bias;
            lds_h1A[sb + ((r + radd) << 4)] = (_Float16)(v > 0.f ? v : 0.f);
        }
    }
    __syncthreads();

    // ---- layer 2: 128 -> 128 ----
    v8f acc2 = {};
    #pragma unroll
    for (int kt = 0; kt < 4; ++kt) {
        v16h a  = *(const v16h*)&lds_h1A[(kt * 32 + lane) << 4];
        v16h bm = *(const v16h*)&W2p[((kt * 8 + wave) * 32 + lane) << 4];
        acc2 = wmma16(a, bm, acc2);
    }
    {
        float bias = b2[nc];
        #pragma unroll
        for (int r = 0; r < 8; ++r) {
            float v = acc2[r] + bias;
            lds_h2A[sb + ((r + radd) << 4)] = (_Float16)(v > 0.f ? v : 0.f);
        }
    }
    __syncthreads();

    // ---- layer 3: 128 -> 14 (padded 16), wave 0 ----
    if (wave == 0) {
        const int n = lane & 15;
        v8f acc3 = {};
        #pragma unroll
        for (int kt = 0; kt < 4; ++kt) {
            v16h a  = *(const v16h*)&lds_h2A[(kt * 32 + lane) << 4];
            v16h bm = *(const v16h*)&W3p[(kt * 32 + lane) << 4];   // ntiles=1
            acc3 = wmma16(a, bm, acc3);
        }
        if (n < NODE_C) {
            float bias = b3[n];
            #pragma unroll
            for (int r = 0; r < 8; ++r)
                out_node[(row0 + r + radd) * NODE_C + n] = sigmoidf_(acc3[r] + bias);
        }
    }
}

// ---------------- Edge MLP: 16 -> relu128 -> relu128 -> sigmoid10 (+pri bit) ----------------
__global__ __launch_bounds__(256) void edge_mlp_kernel(
    const float* __restrict__ bbox, const float* __restrict__ dirs,
    const float* __restrict__ pri,
    const _Float16* __restrict__ W1p, const float* __restrict__ b1,
    const _Float16* __restrict__ W2p, const float* __restrict__ b2,
    const _Float16* __restrict__ W3p, const float* __restrict__ b3,
    float* __restrict__ out_edge,    // [NPAIR, 11]
    float* __restrict__ out_dense)   // [B,N,N,11]
{
    __shared__ _Float16 lds_pA[512];          // 1 K-tile of pair-feature A fragments
    __shared__ _Float16 lds_h1A[16 * Hdim];
    __shared__ _Float16 lds_h2A[16 * Hdim];
    __shared__ int   lds_bb[16], lds_ii[16], lds_jj[16];
    __shared__ float lds_hp[16];

    const int row0 = blockIdx.x * 16;
    const int tid  = threadIdx.x;
    const int lane = tid & 31;
    const int wave = tid >> 5;
    const int nc   = wave * 16 + (lane & 15);
    const int sb   = frag_slot0(nc);
    const int radd = (lane >= 16) ? 8 : 0;

    // decode the 16 pair rows once
    if (tid < 16) {
        int rr  = row0 + tid;
        int b   = rr / PAIRS_PER_B;
        int rem = rr % PAIRS_PER_B;
        int i   = rem / (Nn - 1);
        int k   = rem % (Nn - 1);
        int j   = (k < i) ? k : k + 1;
        lds_bb[tid] = b; lds_ii[tid] = i; lds_jj[tid] = j;
        lds_hp[tid] = (pri[b * Nn + i] > pri[b * Nn + j]) ? 1.0f : 0.0f;
    }
    __syncthreads();

    // build the 16x32 (K padded) A fragment tile directly in fragment layout
    for (int s = tid; s < 512; s += 256) {
        int e = s & 15, l = s >> 4;
        int ka = ak_idx(l, e);
        int mrow = l & 15;
        float v = 0.f;
        if (ka < 16) {
            int b = lds_bb[mrow];
            int src = (ka < 8) ? lds_ii[mrow] : lds_jj[mrow];
            int ff = ka & 7;
            v = (ff < 4) ? bbox[(b * Nn + src) * 4 + ff] * (1.0f / 1024.0f)
                         : dirs[(b * Nn + src) * 4 + (ff - 4)];
        }
        lds_pA[s] = (_Float16)v;
    }
    __syncthreads();

    // ---- layer 1: one WMMA per wave ----
    v8f acc = {};
    {
        v16h a  = *(const v16h*)&lds_pA[lane << 4];
        v16h bm = *(const v16h*)&W1p[(wave * 32 + lane) << 4];
        acc = wmma16(a, bm, acc);
    }
    {
        float bias = b1[nc];
        #pragma unroll
        for (int r = 0; r < 8; ++r) {
            float v = acc[r] + bias;
            lds_h1A[sb + ((r + radd) << 4)] = (_Float16)(v > 0.f ? v : 0.f);
        }
    }
    __syncthreads();

    // ---- layer 2: 128 -> 128 ----
    v8f acc2 = {};
    #pragma unroll
    for (int kt = 0; kt < 4; ++kt) {
        v16h a  = *(const v16h*)&lds_h1A[(kt * 32 + lane) << 4];
        v16h bm = *(const v16h*)&W2p[((kt * 8 + wave) * 32 + lane) << 4];
        acc2 = wmma16(a, bm, acc2);
    }
    {
        float bias = b2[nc];
        #pragma unroll
        for (int r = 0; r < 8; ++r) {
            float v = acc2[r] + bias;
            lds_h2A[sb + ((r + radd) << 4)] = (_Float16)(v > 0.f ? v : 0.f);
        }
    }
    __syncthreads();

    // ---- layer 3: 128 -> 10 (padded 16) + priority bit; wave 0 writes outputs ----
    if (wave == 0) {
        const int n = lane & 15;
        v8f acc3 = {};
        #pragma unroll
        for (int kt = 0; kt < 4; ++kt) {
            v16h a  = *(const v16h*)&lds_h2A[(kt * 32 + lane) << 4];
            v16h bm = *(const v16h*)&W3p[(kt * 32 + lane) << 4];   // ntiles=1
            acc3 = wmma16(a, bm, acc3);
        }
        if (n < EDGE_C + 1) {
            float bias = (n < EDGE_C) ? b3[n] : 0.f;
            #pragma unroll
            for (int r = 0; r < 8; ++r) {
                int mr = r + radd;
                int b = lds_bb[mr], i = lds_ii[mr], j = lds_jj[mr];
                float v = (n < EDGE_C) ? sigmoidf_(acc3[r] + bias) : lds_hp[mr];
                out_edge[(row0 + mr) * (EDGE_C + 1) + n] = v;
                out_dense[((b * Nn + i) * Nn + j) * (EDGE_C + 1) + n] = v;
            }
        }
    }
}

extern "C" void kernel_launch(void* const* d_in, const int* in_sizes, int n_in,
                              void* d_out, int out_size, void* d_ws, size_t ws_size,
                              hipStream_t stream) {
    const float* roi  = (const float*)d_in[0];
    const float* bbox = (const float*)d_in[1];
    const float* dirs = (const float*)d_in[2];
    const float* pri  = (const float*)d_in[3];
    const float* nW1  = (const float*)d_in[4];
    const float* nb1  = (const float*)d_in[5];
    const float* nW2  = (const float*)d_in[6];
    const float* nb2  = (const float*)d_in[7];
    const float* nW3  = (const float*)d_in[8];
    const float* nb3  = (const float*)d_in[9];
    const float* eW1  = (const float*)d_in[10];
    const float* eb1  = (const float*)d_in[11];
    const float* eW2  = (const float*)d_in[12];
    const float* eb2  = (const float*)d_in[13];
    const float* eW3  = (const float*)d_in[14];
    const float* eb3  = (const float*)d_in[15];

    float* out       = (float*)d_out;
    float* out_node  = out;
    float* out_edge  = out + NODE_OUT;
    float* out_dense = out + NODE_OUT + EDGE_ATTR_OUT;

    // f16 fragment-packed weights in workspace (~606 KB total)
    _Float16* ws   = (_Float16*)d_ws;
    _Float16* nW1p = ws;                         // 64*8*512
    _Float16* nW2p = nW1p + 64 * 8 * 512;        // 4*8*512
    _Float16* nW3p = nW2p + 4 * 8 * 512;         // 4*1*512
    _Float16* eW1p = nW3p + 4 * 1 * 512;         // 1*8*512
    _Float16* eW2p = eW1p + 1 * 8 * 512;         // 4*8*512
    _Float16* eW3p = eW2p + 4 * 8 * 512;         // 4*1*512

    // 1) pack weights (once per launch; L2-resident thereafter)
    pack_w_kernel<<<(64 * 8 * 512 + 255) / 256, 256, 0, stream>>>(nW1, CIMG, Hdim, 64, 8, nW1p);
    pack_w_kernel<<<( 4 * 8 * 512 + 255) / 256, 256, 0, stream>>>(nW2, Hdim, Hdim,  4, 8, nW2p);
    pack_w_kernel<<<( 4 * 1 * 512 + 255) / 256, 256, 0, stream>>>(nW3, Hdim, NODE_C, 4, 1, nW3p);
    pack_w_kernel<<<( 1 * 8 * 512 + 255) / 256, 256, 0, stream>>>(eW1, 16,   Hdim,  1, 8, eW1p);
    pack_w_kernel<<<( 4 * 8 * 512 + 255) / 256, 256, 0, stream>>>(eW2, Hdim, Hdim,  4, 8, eW2p);
    pack_w_kernel<<<( 4 * 1 * 512 + 255) / 256, 256, 0, stream>>>(eW3, Hdim, EDGE_C, 4, 1, eW3p);

    // 2) zero the dense relation tensor (diagonal stays zero; d_out is poisoned)
    int n4 = DENSE_OUT / 4;
    zero_f4_kernel<<<(n4 + 255) / 256, 256, 0, stream>>>((float4*)out_dense, n4);

    // 3) node MLP: 2048 rows -> 128 M-tiles
    node_mlp_kernel<<<(Bsz * Nn) / 16, 256, 0, stream>>>(
        roi, nW1p, nb1, nW2p, nb2, nW3p, nb3, out_node);

    // 4) edge MLP: 522240 pair rows -> 32640 M-tiles
    edge_mlp_kernel<<<NPAIR / 16, 256, 0, stream>>>(
        bbox, dirs, pri, eW1p, eb1, eW2p, eb2, eW3p, eb3, out_edge, out_dense);
}
